// NeuralFieldDiffusion_824633720910
// MI455X (gfx1250) — matrix operand order repaired and checked by reading, not compiled, not run on hardware
//
#include <hip/hip_runtime.h>
#include <hip/hip_bf16.h>
#include <math.h>

typedef __bf16 bf16;
typedef __bf16 v8bf  __attribute__((ext_vector_type(8)));
typedef __bf16 v16bf __attribute__((ext_vector_type(16)));
typedef float  v8f   __attribute__((ext_vector_type(8)));

#define HID 1024
#define HEADS 16
#define HD 64
#define ROPE_HALF 30        // ROPE_DIM/2 = 30 rotation pairs
#define NB 2
#define SEQ 2048
#define TOK (NB*SEQ)
#define SWI 2730
#define SWI_PAD 2752        // next multiple of 32 (WMMA K granularity)
#define EPSF 1e-6f

static __device__ __forceinline__ v16bf cat8(v8bf lo, v8bf hi) {
  return __builtin_shufflevector(lo, hi, 0,1,2,3,4,5,6,7,8,9,10,11,12,13,14,15);
}
static __device__ __forceinline__ v8f wmma_bf16(v16bf a, v16bf b, v8f c) {
  return __builtin_amdgcn_wmma_f32_16x16x32_bf16(false, a, false, b, (short)0, c, false, false);
}
// Async global->LDS 16B copy (per-lane), tracked by ASYNCcnt.
static __device__ __forceinline__ void async_b128(unsigned lds_off, const bf16* g) {
  asm volatile("global_load_async_to_lds_b128 %0, %1, off"
               :: "v"(lds_off), "v"(g) : "memory");
}
static __device__ __forceinline__ unsigned lds_off_u32(const void* p) {
  // generic pointers to LDS carry the wave-relative LDS byte offset in the low 32 bits
  return (unsigned)(uintptr_t)p;
}

// ---------------- weight conversion ----------------
__global__ __launch_bounds__(256) void conv_bf16(const float* __restrict__ src,
                                                 bf16* __restrict__ dst, int n) {
  int i = blockIdx.x*256 + threadIdx.x;
  if (i < n) dst[i] = (bf16)src[i];
}
__global__ __launch_bounds__(256) void conv_pad_w2(const float* __restrict__ src,
                                                   bf16* __restrict__ dst) {
  int i = blockIdx.x*256 + threadIdx.x;
  if (i >= HID*SWI_PAD) return;
  int o = i / SWI_PAD, kk = i % SWI_PAD;
  dst[i] = (kk < SWI) ? (bf16)src[(size_t)o*SWI + kk] : (bf16)0.f;
}

// ---------------- adaLN modulation: mod = c @ ada_w.T + ada_b ----------------
__global__ __launch_bounds__(256) void ada_gemv(const float* __restrict__ c,
                                                const float* __restrict__ aw,
                                                const float* __restrict__ ab,
                                                float* __restrict__ mod) {
  int idx = blockIdx.x*256 + threadIdx.x;
  if (idx >= NB*6*HID) return;
  int b = idx / (6*HID), oo = idx % (6*HID);
  const float* cr = c + (size_t)b*HID;
  const float* wr = aw + (size_t)oo*HID;
  float s = 0.f;
  for (int kk = 0; kk < HID; ++kk) s += cr[kk]*wr[kk];
  mod[idx] = s + ab[oo];
}

// ---------------- fused RMSNorm + modulate -> bf16 ----------------
__global__ __launch_bounds__(256) void norm_mod(const float* __restrict__ x,
                                                const float* __restrict__ w,
                                                const float* __restrict__ mod,
                                                int sh_off, int sc_off,
                                                bf16* __restrict__ out) {
  __shared__ float red[8];
  int tok = blockIdx.x;
  int b = tok / SEQ;
  const float* xr = x + (size_t)tok*HID;
  float vals[4], loc = 0.f;
#pragma unroll
  for (int i = 0; i < 4; ++i) { int cc = threadIdx.x + i*256; vals[i] = xr[cc]; loc += vals[i]*vals[i]; }
#pragma unroll
  for (int m = 16; m >= 1; m >>= 1) loc += __shfl_xor(loc, m, 32);
  if ((threadIdx.x & 31) == 0) red[threadIdx.x >> 5] = loc;
  __syncthreads();
  float tot = 0.f;
#pragma unroll
  for (int i = 0; i < 8; ++i) tot += red[i];
  float r = rsqrtf(tot*(1.f/HID) + EPSF);
  const float* mb = mod + (size_t)b*6*HID;
#pragma unroll
  for (int i = 0; i < 4; ++i) {
    int cc = threadIdx.x + i*256;
    float sh = mb[sh_off + cc], sc = mb[sc_off + cc];
    out[(size_t)tok*HID + cc] = (bf16)((vals[i]*r*w[cc])*(1.f + sc) + sh);
  }
}

// ---------------- WMMA GEMM with async double-buffered LDS staging ----------------
// out[M,O] = A[M,K](bf16) x W[O,K](bf16)^T.  Block tile 64x128, 8 waves (4Mx2N),
// K step 32.  Tiles staged via GLOBAL_LOAD_ASYNC_TO_LDS_B128 (ASYNCcnt), fragments
// fed from LDS; next stage's DMA overlaps current stage's WMMAs.
__global__ __launch_bounds__(256) void gemm_wmma(const bf16* __restrict__ A, int lda,
                                                 const bf16* __restrict__ W, int ldw,
                                                 float* __restrict__ outF,
                                                 bf16* __restrict__ outH,
                                                 int K, int O, int ldc) {
  __shared__ __align__(16) bf16 sA[2][64*32];   // 2 x 4 KB
  __shared__ __align__(16) bf16 sB[2][128*32];  // 2 x 8 KB
  const int tid  = threadIdx.x;
  const int wid  = tid >> 5;
  const int lane = tid & 31;
  const int half = lane >> 4, lm = lane & 15;
  const int mw = wid & 3, nw = wid >> 2;
  const int mblk = blockIdx.y*64, nblk = blockIdx.x*128;
  const int klo = half*8;

  // staging assignment: A 64 rows x 4 chunks(16B); B 128 rows x 2 chunks(32B)
  const int arow = tid >> 2, achk = tid & 3;
  const int brow = tid >> 1, bchk = tid & 1;
  const bf16* Ag = A + (size_t)(mblk + arow)*lda + achk*8;
  int brc = nblk + brow; brc = brc < O ? brc : O - 1;   // clamp (stores guarded)
  const bf16* Bg = W + (size_t)brc*ldw + bchk*16;

  const unsigned saoff = lds_off_u32(&sA[0][arow*32 + achk*8]);
  const unsigned sboff = lds_off_u32(&sB[0][brow*32 + bchk*16]);
  const unsigned abuf  = (unsigned)(64*32*2);   // bytes between A buffers
  const unsigned bbuf  = (unsigned)(128*32*2);  // bytes between B buffers

  auto stage = [&](int buf, int k0) {
    async_b128(saoff + (unsigned)buf*abuf, Ag + k0);
    async_b128(sboff + (unsigned)buf*bbuf, Bg + k0);
    async_b128(sboff + (unsigned)buf*bbuf + 16u, Bg + k0 + 8);
  };

  v8f acc[4] = {};
  const int iters = K >> 5;
  stage(0, 0);
  for (int i = 0; i < iters; ++i) {
    asm volatile("s_wait_asynccnt 0" ::: "memory");   // stage i landed (this wave)
    __syncthreads();                                  // landed for all waves
    if (i + 1 < iters) stage((i + 1) & 1, (i + 1) << 5);  // overlap next DMA
    const int buf = i & 1;
    const bf16* aT = &sA[buf][(mw*16 + lm)*32];
    v16bf a = cat8(*(const v8bf*)(aT + klo), *(const v8bf*)(aT + 16 + klo));
#pragma unroll
    for (int f = 0; f < 4; ++f) {
      v16bf b = cat8(*(const v8bf*)&sB[buf][(nw*64 + f*16 + lm)*32 + klo], *(const v8bf*)&sB[buf][(nw*64 + f*16 + lm)*32 + 16 + klo]);
      acc[f] = wmma_bf16(a, b, acc[f]);
    }
    __syncthreads();                                  // reads done before rewrite
  }
#pragma unroll
  for (int f = 0; f < 4; ++f) {
    int col = nblk + nw*64 + f*16 + lm;
    if (col >= O) continue;
#pragma unroll
    for (int v = 0; v < 8; ++v) {
      int row = mblk + mw*16 + v + half*8;
      if (outF) outF[(size_t)row*ldc + col] = acc[f][v];
      else      outH[(size_t)row*ldc + col] = (bf16)acc[f][v];
    }
  }
}

// ---------------- per-head RMSNorm(q,k) + 3D RoPE + V transpose ----------------
__global__ __launch_bounds__(64) void qkv_post(const bf16* __restrict__ qkv,
                                               const float* __restrict__ qw,
                                               const float* __restrict__ kw,
                                               const float* __restrict__ pos,
                                               bf16* __restrict__ q,
                                               bf16* __restrict__ k,
                                               bf16* __restrict__ vT) {
  int tok = blockIdx.x; int b = tok / SEQ; int n = tok % SEQ;
  int t = threadIdx.x; if (t >= 48) return;
  int role = t >> 4, h = t & 15;
  const bf16* src = qkv + ((size_t)tok*3 + role)*HID + h*HD;
  if (role == 2) {                               // V: transpose to [B,H,D,N]
    bf16* dst = vT + (size_t)(b*HEADS + h)*HD*SEQ + n;
    for (int d = 0; d < HD; ++d) dst[(size_t)d*SEQ] = src[d];
    return;
  }
  float v[HD]; float ss = 0.f;
  for (int d = 0; d < HD; ++d) { v[d] = (float)src[d]; ss += v[d]*v[d]; }
  float r = rsqrtf(ss*(1.f/HD) + EPSF);
  const float* w = role ? kw : qw;
  for (int d = 0; d < HD; ++d) v[d] *= r*w[d];
  float p3[3];
  for (int a = 0; a < 3; ++a) p3[a] = pos[(size_t)tok*3 + a];
  for (int i = 0; i < ROPE_HALF; ++i) {
    int axis = i / 10, j = i % 10;
    float freq = __expf(-(float)j * (9.210340371976184f / 10.f));  // theta^{-j/10}
    float ang = p3[axis]*freq;
    float cs = __cosf(ang), sn = __sinf(ang);
    float t0 = v[2*i], t1 = v[2*i+1];
    v[2*i]   = t0*cs - t1*sn;
    v[2*i+1] = t0*sn + t1*cs;
  }
  bf16* dst = (role ? k : q) + ((size_t)(b*HEADS + h)*SEQ + n)*HD;
  for (int d = 0; d < HD; ++d) dst[d] = (bf16)v[d];
}

// ---------------- flash attention: one wave per 16 query rows ----------------
__global__ __launch_bounds__(128) void attn_flash(const bf16* __restrict__ q,
                                                  const bf16* __restrict__ k,
                                                  const bf16* __restrict__ vT,
                                                  bf16* __restrict__ o) {
  __shared__ __align__(16) bf16 plds[4*16*32];
  int wid  = threadIdx.x >> 5;
  int lane = threadIdx.x & 31;
  int half = lane >> 4, lm = lane & 15;
  int b = blockIdx.z, h = blockIdx.y;
  int m0 = blockIdx.x*64 + wid*16;
  const bf16* qh = q  + (size_t)(b*HEADS + h)*SEQ*HD;
  const bf16* kh = k  + (size_t)(b*HEADS + h)*SEQ*HD;
  const bf16* vh = vT + (size_t)(b*HEADS + h)*HD*SEQ;   // [D, N]
  bf16* P = plds + wid*512;
  int klo = half*8;
  const bf16* qp = qh + (size_t)(m0 + lm)*HD + half*16;
  v16bf aq0 = *(const v16bf*)(qp);        // d 0..31
  v16bf aq1 = *(const v16bf*)(qp + 32);   // d 32..63
  v8f oacc[4] = {};
  float ms[8], ls[8];
#pragma unroll
  for (int v = 0; v < 8; ++v) { ms[v] = -1e30f; ls[v] = 0.f; }
  const float scale = 0.125f;  // 1/sqrt(64)
  for (int j0 = 0; j0 < SEQ; j0 += 32) {
    v8f s0 = {}, s1 = {};
    const bf16* kp0 = kh + (size_t)(j0 + lm)*HD      + half*16;
    const bf16* kp1 = kh + (size_t)(j0 + 16 + lm)*HD + half*16;
    s0 = wmma_bf16(aq0, *(const v16bf*)(kp0),      s0);
    s0 = wmma_bf16(aq1, *(const v16bf*)(kp0 + 32), s0);
    s1 = wmma_bf16(aq0, *(const v16bf*)(kp1),      s1);
    s1 = wmma_bf16(aq1, *(const v16bf*)(kp1 + 32), s1);
    float p0[8], p1[8];
#pragma unroll
    for (int v = 0; v < 8; ++v) {
      float a0 = s0[v]*scale, a1 = s1[v]*scale;
      float tm = fmaxf(a0, a1);
#pragma unroll
      for (int m = 8; m >= 1; m >>= 1) tm = fmaxf(tm, __shfl_xor(tm, m, 32)); // stays in half
      float nm = fmaxf(ms[v], tm);
      float al = __expf(ms[v] - nm); ms[v] = nm;
      p0[v] = __expf(a0 - nm); p1[v] = __expf(a1 - nm);
      float rs = p0[v] + p1[v];
#pragma unroll
      for (int m = 8; m >= 1; m >>= 1) rs += __shfl_xor(rs, m, 32);
      ls[v] = ls[v]*al + rs;
      oacc[0][v] *= al; oacc[1][v] *= al; oacc[2][v] *= al; oacc[3][v] *= al;
    }
    // C-layout -> A-layout via LDS (per-wave 16x32 bf16 tile)
#pragma unroll
    for (int v = 0; v < 8; ++v) {
      P[(v + 8*half)*32 + lm]      = (bf16)p0[v];
      P[(v + 8*half)*32 + 16 + lm] = (bf16)p1[v];
    }
    asm volatile("s_wait_dscnt 0" ::: "memory");
    v16bf ap = cat8(*(const v8bf*)(P + lm*32 + klo),
                    *(const v8bf*)(P + lm*32 + 16 + klo));
#pragma unroll
    for (int f = 0; f < 4; ++f) {
      v16bf bv = cat8(*(const v8bf*)(vh + (size_t)(f*16 + lm)*SEQ + j0 + klo), *(const v8bf*)(vh + (size_t)(f*16 + lm)*SEQ + j0 + 16 + klo));
      oacc[f] = wmma_bf16(ap, bv, oacc[f]);
    }
  }
#pragma unroll
  for (int f = 0; f < 4; ++f)
#pragma unroll
    for (int v = 0; v < 8; ++v) {
      int tokrow = m0 + v + 8*half;
      float val = oacc[f][v] / ls[v];
      o[((size_t)(b*SEQ + tokrow)*HEADS + h)*HD + f*16 + lm] = (bf16)val;  // [B,N,H,D]
    }
}

// ---------------- epilogues ----------------
__global__ __launch_bounds__(256) void resid_msa(const float* __restrict__ x,
                                                 const float* __restrict__ pb,
                                                 const float* __restrict__ mod,
                                                 float* __restrict__ x1) {
  int i = blockIdx.x*256 + threadIdx.x;
  if (i >= TOK*HID) return;
  int tok = i / HID, cc = i % HID, b = tok / SEQ;
  float g = mod[(size_t)b*6*HID + 2*HID + cc];
  x1[i] = x[i] + g*(x1[i] + pb[cc]);
}
__global__ __launch_bounds__(256) void silu_mul(bf16* __restrict__ a,
                                                const bf16* __restrict__ bb) {
  int i = blockIdx.x*256 + threadIdx.x;
  if (i >= TOK*SWI_PAD) return;
  int cc = i % SWI_PAD;
  if (cc >= SWI) { a[i] = (bf16)0.f; return; }  // zero pad cols for w2 GEMM
  float av = (float)a[i], bv = (float)bb[i];
  a[i] = (bf16)((av / (1.f + __expf(-av))) * bv);
}
__global__ __launch_bounds__(256) void resid_mlp(const float* __restrict__ x1,
                                                 const float* __restrict__ mod,
                                                 float* __restrict__ out) {
  int i = blockIdx.x*256 + threadIdx.x;
  if (i >= TOK*HID) return;
  int tok = i / HID, cc = i % HID, b = tok / SEQ;
  float g = mod[(size_t)b*6*HID + 5*HID + cc];
  out[i] = x1[i] + g*out[i];
}

extern "C" void kernel_launch(void* const* d_in, const int* in_sizes, int n_in,
                              void* d_out, int out_size, void* d_ws, size_t ws_size,
                              hipStream_t stream) {
  const float* x    = (const float*)d_in[0];
  const float* c    = (const float*)d_in[1];
  const float* pos  = (const float*)d_in[2];
  const float* n1w  = (const float*)d_in[3];
  const float* qkvw = (const float*)d_in[4];
  const float* qnw  = (const float*)d_in[5];
  const float* knw  = (const float*)d_in[6];
  const float* pw   = (const float*)d_in[7];
  const float* pb   = (const float*)d_in[8];
  const float* n2w  = (const float*)d_in[9];
  const float* w1   = (const float*)d_in[10];
  const float* w3   = (const float*)d_in[11];
  const float* w2   = (const float*)d_in[12];
  const float* aw   = (const float*)d_in[13];
  const float* ab   = (const float*)d_in[14];
  float* out = (float*)d_out;

  char* ws = (char*)d_ws;
  size_t off = 0;
  auto alloc = [&](size_t bytes) { void* p = ws + off; off += (bytes + 255) & ~(size_t)255; return p; };
  float* mod    = (float*)alloc((size_t)NB*6*HID*4);
  bf16* wb_qkv  = (bf16*)alloc((size_t)3*HID*HID*2);
  bf16* wb_proj = (bf16*)alloc((size_t)HID*HID*2);
  bf16* wb_w1   = (bf16*)alloc((size_t)SWI*HID*2);
  bf16* wb_w3   = (bf16*)alloc((size_t)SWI*HID*2);
  bf16* wb_w2   = (bf16*)alloc((size_t)HID*SWI_PAD*2);
  bf16* hbuf    = (bf16*)alloc((size_t)TOK*HID*2);      // h, reused as h2
  bf16* qkv     = (bf16*)alloc((size_t)TOK*3*HID*2);    // reused as ff_a (TOK*SWI_PAD*2 fits)
  bf16* qbuf    = (bf16*)alloc((size_t)TOK*HID*2);      // q..vT contiguous; reused as ff_b
  bf16* kbuf    = (bf16*)alloc((size_t)TOK*HID*2);
  bf16* vtbuf   = (bf16*)alloc((size_t)TOK*HID*2);
  bf16* obuf    = (bf16*)alloc((size_t)TOK*HID*2);
  float* x1     = (float*)alloc((size_t)TOK*HID*4);
  bf16* ffa = qkv;
  bf16* ffb = qbuf;

  conv_bf16<<<(3*HID*HID + 255)/256, 256, 0, stream>>>(qkvw, wb_qkv, 3*HID*HID);
  conv_bf16<<<(HID*HID + 255)/256,   256, 0, stream>>>(pw,   wb_proj, HID*HID);
  conv_bf16<<<(SWI*HID + 255)/256,   256, 0, stream>>>(w1,   wb_w1,   SWI*HID);
  conv_bf16<<<(SWI*HID + 255)/256,   256, 0, stream>>>(w3,   wb_w3,   SWI*HID);
  conv_pad_w2<<<(HID*SWI_PAD + 255)/256, 256, 0, stream>>>(w2, wb_w2);

  ada_gemv<<<(NB*6*HID + 255)/256, 256, 0, stream>>>(c, aw, ab, mod);
  norm_mod<<<TOK, 256, 0, stream>>>(x, n1w, mod, 0, HID, hbuf);

  dim3 g1(3*HID/128, TOK/64);
  gemm_wmma<<<g1, 256, 0, stream>>>(hbuf, HID, wb_qkv, HID, nullptr, qkv, HID, 3*HID, 3*HID);

  qkv_post<<<TOK, 64, 0, stream>>>(qkv, qnw, knw, pos, qbuf, kbuf, vtbuf);

  dim3 ga(SEQ/64, HEADS, NB);
  attn_flash<<<ga, 128, 0, stream>>>(qbuf, kbuf, vtbuf, obuf);

  dim3 g2(HID/128, TOK/64);
  gemm_wmma<<<g2, 256, 0, stream>>>(obuf, HID, wb_proj, HID, x1, nullptr, HID, HID, HID);
  resid_msa<<<(TOK*HID + 255)/256, 256, 0, stream>>>(x, pb, mod, x1);

  norm_mod<<<TOK, 256, 0, stream>>>(x1, n2w, mod, 3*HID, 4*HID, hbuf);

  dim3 g3((SWI + 127)/128, TOK/64);
  gemm_wmma<<<g3, 256, 0, stream>>>(hbuf, HID, wb_w1, HID, nullptr, ffa, HID, SWI, SWI_PAD);
  gemm_wmma<<<g3, 256, 0, stream>>>(hbuf, HID, wb_w3, HID, nullptr, ffb, HID, SWI, SWI_PAD);
  silu_mul<<<(TOK*SWI_PAD + 255)/256, 256, 0, stream>>>(ffa, ffb);

  dim3 g4(HID/128, TOK/64);
  gemm_wmma<<<g4, 256, 0, stream>>>(ffa, SWI_PAD, wb_w2, SWI_PAD, out, nullptr, SWI_PAD, HID, HID);
  resid_mlp<<<(TOK*HID + 255)/256, 256, 0, stream>>>(x1, mod, out);
}
